// BiRNN_IFBUS3S8_84636625535335
// MI455X (gfx1250) — matrix operand order repaired and checked by reading, not compiled, not run on hardware
//
#include <hip/hip_runtime.h>
#include <hip/hip_bf16.h>
#include <math.h>

// ---------------------------------------------------------------------------
// Types for CDNA5 WMMA (gfx1250, wave32)
// ---------------------------------------------------------------------------
typedef __bf16 bf16_t;
typedef bf16_t v16bf __attribute__((ext_vector_type(16)));
typedef float  v8f   __attribute__((ext_vector_type(8)));
typedef unsigned int u32x8 __attribute__((ext_vector_type(8)));

#define DEV static __device__ __forceinline__

DEV unsigned short f2bf(float f) {
  unsigned u = __builtin_bit_cast(unsigned, f);
  u += 0x7FFFu + ((u >> 16) & 1u);            // round-to-nearest-even
  return (unsigned short)(u >> 16);
}
DEV unsigned pack2bf(float lo, float hi) {
  return (unsigned)f2bf(lo) | ((unsigned)f2bf(hi) << 16);
}
DEV v16bf as_v16bf(u32x8 u) { return __builtin_bit_cast(v16bf, u); }

DEV v8f wmma_bf16(v16bf a, v16bf b, v8f c) {
  // D = A(16x32) * B(32x16) + C(16x16 f32)
  return __builtin_amdgcn_wmma_f32_16x16x32_bf16(
      false, a, false, b, (short)0, c, false, false);
}
DEV float sigm(float x) { return 1.0f / (1.0f + __expf(-x)); }

// ---------------------------------------------------------------------------
// Pack a row-major fp32 [M,K] matrix into A-fragment layout:
//   dst dword index = ((mtile*ksteps + ks)*32 + lane)*8 + p
//   A 16-bit 16x32 layout: lane<16: K = {0..7}U{16..23}; lane>=16: +8
// ---------------------------------------------------------------------------
__global__ void pack_a_kernel(const float* __restrict__ src,
                              unsigned* __restrict__ dst,
                              int ksteps, int K, int relu, int total) {
  int idx = blockIdx.x * blockDim.x + threadIdx.x;
  if (idx >= total) return;
  int p     = idx & 7;
  int lane  = (idx >> 3) & 31;
  int ks    = (idx >> 8) % ksteps;
  int mtile = (idx >> 8) / ksteps;
  int row = mtile * 16 + (lane & 15);
  int k0  = ks * 32 + ((p >= 4) ? 16 : 0) + ((lane & 16) ? 8 : 0) + ((p & 3) << 1);
  size_t s = (size_t)row * K + k0;
  float a = src[s], b = src[s + 1];
  if (relu) { a = fmaxf(a, 0.f); b = fmaxf(b, 0.f); }
  dst[idx] = pack2bf(a, b);
}

// ---------------------------------------------------------------------------
// Pack a row-major fp32 [K,N] matrix into B-fragment layout:
//   B 16-bit 32x16 layout: lane<16: K = {0..7}U{16..23}; lane>=16: +8
// ---------------------------------------------------------------------------
__global__ void pack_b_kernel(const float* __restrict__ src,
                              unsigned* __restrict__ dst,
                              int ksteps, int N, int total) {
  int idx = blockIdx.x * blockDim.x + threadIdx.x;
  if (idx >= total) return;
  int p     = idx & 7;
  int lane  = (idx >> 3) & 31;
  int ks    = (idx >> 8) % ksteps;
  int ntile = (idx >> 8) / ksteps;
  int n  = ntile * 16 + (lane & 15);
  int k0 = ks * 32 + ((p >= 4) ? 16 : 0) + ((lane & 16) ? 8 : 0) + ((p & 3) << 1);
  size_t s = (size_t)k0 * N + n;
  dst[idx] = pack2bf(src[s], src[s + N]);
}

// ---------------------------------------------------------------------------
// Build combined projection weight [K=1024][N=16] (cols: Wp0,Wp1,Wbu0,Wbu1,
// Wss3 x3, Wss8 x8, pad) and combined bias[16].
// ---------------------------------------------------------------------------
__global__ void build_pw_kernel(const float* __restrict__ Wp,
                                const float* __restrict__ Wbu,
                                const float* __restrict__ Wss3,
                                const float* __restrict__ Wss8,
                                const float* __restrict__ bp,
                                const float* __restrict__ bbu,
                                const float* __restrict__ bss3,
                                const float* __restrict__ bss8,
                                float* __restrict__ Pw,
                                float* __restrict__ pbias) {
  int idx = blockIdx.x * blockDim.x + threadIdx.x;
  if (idx < 16) {
    int n = idx;
    pbias[n] = (n < 2) ? bp[n] : (n < 4) ? bbu[n - 2]
             : (n < 7) ? bss3[n - 4] : (n < 15) ? bss8[n - 7] : 0.f;
  }
  if (idx < 16384) {
    int k = idx >> 4, n = idx & 15;
    float v = (n < 2) ? Wp[k * 2 + n]
            : (n < 4) ? Wbu[k * 2 + (n - 2)]
            : (n < 7) ? Wss3[k * 3 + (n - 4)]
            : (n < 15) ? Wss8[k * 8 + (n - 7)] : 0.f;
    Pw[idx] = v;
  }
}

// ---------------------------------------------------------------------------
// Big GEMM: C[M=32768, N=Ntiles*16] = Apk * Bpk + bias   (bf16 WMMA, f32 acc)
// Block: 256 thr = 8 waves as 4(M)x2(N); wave tile 32x64 (2x4 16x16 tiles).
// ---------------------------------------------------------------------------
__global__ __launch_bounds__(256) void gemm_kernel(
    const u32x8* __restrict__ A, const u32x8* __restrict__ B,
    const float* __restrict__ bias, float* __restrict__ C,
    int ksteps, int Ntiles) {
  const int lane = threadIdx.x & 31;
  const int w    = threadIdx.x >> 5;
  const int wm = w >> 1, wn = w & 1;
  const int mt0 = blockIdx.x * 8 + wm * 2;
  const int nt0 = blockIdx.y * 8 + wn * 4;

  v8f acc[2][4];
#pragma unroll
  for (int i = 0; i < 2; ++i)
#pragma unroll
    for (int j = 0; j < 4; ++j) acc[i][j] = (v8f){0,0,0,0,0,0,0,0};

  float bn[4];
#pragma unroll
  for (int j = 0; j < 4; ++j) bn[j] = bias[(nt0 + j) * 16 + (lane & 15)];

  for (int ks = 0; ks < ksteps; ++ks) {
    v16bf a0 = as_v16bf(A[((size_t)(mt0 + 0) * ksteps + ks) * 32 + lane]);
    v16bf a1 = as_v16bf(A[((size_t)(mt0 + 1) * ksteps + ks) * 32 + lane]);
#pragma unroll
    for (int j = 0; j < 4; ++j) {
      v16bf bm = as_v16bf(B[((size_t)(nt0 + j) * ksteps + ks) * 32 + lane]);
      acc[0][j] = wmma_bf16(a0, bm, acc[0][j]);
      acc[1][j] = wmma_bf16(a1, bm, acc[1][j]);
    }
  }

  const int ldN = Ntiles * 16;
  const int hh = (lane >> 4) << 3;   // 0 or 8
#pragma unroll
  for (int i = 0; i < 2; ++i)
#pragma unroll
    for (int j = 0; j < 4; ++j) {
      int n = (nt0 + j) * 16 + (lane & 15);
      int mbase = (mt0 + i) * 16 + hh;
#pragma unroll
      for (int e = 0; e < 8; ++e)
        C[(size_t)(mbase + e) * ldN + n] = acc[i][j][e] + bn[j];
    }
}

// ---------------------------------------------------------------------------
// Persistent LSTM recurrence. grid = 2 (dir), block = 512 (16 waves).
// h lives in LDS (bf16, padded stride), U is pre-packed B-fragments in L2.
// Wave g owns columns n' in [g*32, g*32+32) of all 4 gate blocks, both mtiles,
// so gates/c/h stay in registers. Keras gate order (i,f,g,o), masking carries
// state AND repeats previous h at masked steps.
// ---------------------------------------------------------------------------
#define HSTR 524   // bf16 row stride (LDS bank-conflict-free for frag gathers)

__global__ __launch_bounds__(512) void lstm_recur_kernel(
    const float* __restrict__ zf, const float* __restrict__ zb,
    const u32x8* __restrict__ Uf, const u32x8* __restrict__ Ub,
    const int* __restrict__ xmask, float* __restrict__ out, int T) {
  __shared__ unsigned short hsh[32 * HSTR];

  const int dir = blockIdx.x;
  const float* __restrict__ Z = dir ? zb : zf;
  const u32x8* __restrict__ U = dir ? Ub : Uf;

  const int lane = threadIdx.x & 31;
  const int g    = threadIdx.x >> 5;      // 0..15: column group
  const int nl   = lane & 15;
  const int hh   = (lane >> 4) << 3;      // 0 or 8

  for (int i = threadIdx.x; i < 32 * HSTR; i += 512) hsh[i] = 0;
  __syncthreads();

  v8f c_st[2][2], h_pv[2][2];
#pragma unroll
  for (int m = 0; m < 2; ++m)
#pragma unroll
    for (int j = 0; j < 2; ++j) {
      c_st[m][j] = (v8f){0,0,0,0,0,0,0,0};
      h_pv[m][j] = (v8f){0,0,0,0,0,0,0,0};
    }

  for (int step = 0; step < T; ++step) {
    const int t = dir ? (T - 1 - step) : step;

    // --- init accumulators with z_t (fuses the z + h@U add) ---------------
    v8f acc[2][4][2];
#pragma unroll
    for (int m = 0; m < 2; ++m)
#pragma unroll
      for (int q = 0; q < 4; ++q)
#pragma unroll
        for (int j = 0; j < 2; ++j) {
          int n = q * 512 + (g * 2 + j) * 16 + nl;
#pragma unroll
          for (int e = 0; e < 8; ++e) {
            int row = m * 16 + hh + e;
            acc[m][q][j][e] = Z[((size_t)row * T + t) * 2048 + n];
          }
        }

    // prefetch next step's z tile into cache hierarchy
    if (step + 1 < T) {
      int t2 = dir ? (T - 2 - step) : (step + 1);
      int prow = threadIdx.x >> 4;             // 0..31
      int pcol = (threadIdx.x & 15) * 128;
#pragma unroll
      for (int pi = 0; pi < 4; ++pi)
        __builtin_prefetch(&Z[((size_t)prow * T + t2) * 2048 + pcol + pi * 32], 0, 1);
    }

    // --- r = h_prev @ U  (bf16 WMMA, K=512 -> 16 ksteps) ------------------
    for (int ks = 0; ks < 16; ++ks) {
      v16bf a[2];
#pragma unroll
      for (int m = 0; m < 2; ++m) {
        u32x8 u;
        int row = m * 16 + nl;
#pragma unroll
        for (int p = 0; p < 8; ++p) {
          int k0 = ks * 32 + ((p >= 4) ? 16 : 0) + ((lane & 16) ? 8 : 0) + ((p & 3) << 1);
          u[p] = *(const unsigned*)&hsh[row * HSTR + k0];
        }
        a[m] = as_v16bf(u);
      }
#pragma unroll
      for (int q = 0; q < 4; ++q)
#pragma unroll
        for (int j = 0; j < 2; ++j) {
          int ntile = q * 32 + g * 2 + j;
          v16bf bm = as_v16bf(U[((size_t)ntile * 16 + ks) * 32 + lane]);
          acc[0][q][j] = wmma_bf16(a[0], bm, acc[0][q][j]);
          acc[1][q][j] = wmma_bf16(a[1], bm, acc[1][q][j]);
        }
    }

    // --- gates (all in registers) -----------------------------------------
#pragma unroll
    for (int m = 0; m < 2; ++m) {
      int vmask = 0;
#pragma unroll
      for (int e = 0; e < 8; ++e) {
        int row = m * 16 + hh + e;
        vmask |= (xmask[row * T + t] != 1) ? (1 << e) : 0;
      }
#pragma unroll
      for (int j = 0; j < 2; ++j)
#pragma unroll
        for (int e = 0; e < 8; ++e) {
          float iv = sigm(acc[m][0][j][e]);
          float fv = sigm(acc[m][1][j][e]);
          float gv = tanhf(acc[m][2][j][e]);
          float ov = sigm(acc[m][3][j][e]);
          float cn = fv * c_st[m][j][e] + iv * gv;
          float hn = ov * tanhf(cn);
          if (vmask & (1 << e)) { c_st[m][j][e] = cn; h_pv[m][j][e] = hn; }
        }
    }

    __syncthreads();   // all waves done reading h LDS of this step

    // --- publish h: LDS (bf16, next step A frags) + global output ---------
#pragma unroll
    for (int m = 0; m < 2; ++m)
#pragma unroll
      for (int j = 0; j < 2; ++j) {
        int n = (g * 2 + j) * 16 + nl;
#pragma unroll
        for (int e = 0; e < 8; ++e) {
          int row = m * 16 + hh + e;
          float hv = h_pv[m][j][e];
          hsh[row * HSTR + n] = f2bf(hv);
          out[((size_t)row * T + t) * 1024 + dir * 512 + n] = hv;
        }
      }

    __syncthreads();   // h visible before next step's A-frag gathers
  }
}

// ---------------------------------------------------------------------------
// Output heads: [32768x1024] x [1024x16] WMMA GEMM; epilogue does the 2-way
// softmax (ppi) via __shfl_xor(.,1,32) and scatters into 4 output regions.
// ---------------------------------------------------------------------------
__global__ __launch_bounds__(256) void proj_kernel(
    const u32x8* __restrict__ A, const u32x8* __restrict__ B,
    const float* __restrict__ bias, float* __restrict__ out) {
  const int lane = threadIdx.x & 31;
  const int w    = threadIdx.x >> 5;
  const int mt   = blockIdx.x * 16 + w * 2;

  v8f acc[2] = {(v8f){0,0,0,0,0,0,0,0}, (v8f){0,0,0,0,0,0,0,0}};
  for (int ks = 0; ks < 32; ++ks) {
    v16bf bm = as_v16bf(B[(size_t)ks * 32 + lane]);
#pragma unroll
    for (int m = 0; m < 2; ++m) {
      v16bf a = as_v16bf(A[((size_t)(mt + m) * 32 + ks) * 32 + lane]);
      acc[m] = wmma_bf16(a, bm, acc[m]);
    }
  }

  const int n  = lane & 15;
  const int hh = (lane >> 4) << 3;
  const float bn = bias[n];
#pragma unroll
  for (int m = 0; m < 2; ++m)
#pragma unroll
    for (int e = 0; e < 8; ++e) {
      int row = (mt + m) * 16 + hh + e;     // row = b*T + t
      float v = acc[m][e] + bn;
      float other = __shfl_xor(v, 1, 32);   // wave32 partner within n-pair
      if (n < 2) {                          // ppi: softmax over 2 logits
        float mx = fmaxf(v, other);
        float e0 = __expf(v - mx), e1 = __expf(other - mx);
        out[row * 2 + n] = e0 / (e0 + e1);
      } else if (n < 4) {
        out[65536 + row * 2 + (n - 2)] = v;       // buried
      } else if (n < 7) {
        out[131072 + row * 3 + (n - 4)] = v;      // ss3
      } else if (n < 15) {
        out[229376 + row * 8 + (n - 7)] = v;      // ss8
      }
    }
}

// ---------------------------------------------------------------------------
// Host orchestration (all launches on `stream`; deterministic, scratch-only)
// ---------------------------------------------------------------------------
extern "C" void kernel_launch(void* const* d_in, const int* in_sizes, int n_in,
                              void* d_out, int out_size, void* d_ws, size_t ws_size,
                              hipStream_t stream) {
  const float* x     = (const float*)d_in[0];
  const int*   xmask = (const int*)  d_in[1];
  const float* Wf    = (const float*)d_in[2];
  const float* Uf    = (const float*)d_in[3];
  const float* bfv   = (const float*)d_in[4];
  const float* Wb    = (const float*)d_in[5];
  const float* Ub    = (const float*)d_in[6];
  const float* bbv   = (const float*)d_in[7];
  const float* Wp    = (const float*)d_in[8];
  const float* bp    = (const float*)d_in[9];
  const float* Wbu   = (const float*)d_in[10];
  const float* bbu   = (const float*)d_in[11];
  const float* Wss3  = (const float*)d_in[12];
  const float* bss3  = (const float*)d_in[13];
  const float* Wss8  = (const float*)d_in[14];
  const float* bss8  = (const float*)d_in[15];

  const int T = 1024;
  const size_t APK_DW = 2048ull * 32 * 256;   // 16,777,216 dwords (64 MB)
  const size_t WPK_DW = 128ull * 32 * 256;    //  1,048,576 dwords ( 4 MB)
  const size_t UPK_DW = 128ull * 16 * 256;    //    524,288 dwords ( 2 MB)
  const size_t Z_FLOATS   = 32768ull * 2048;  // 256 MB
  const size_t SEQ_FLOATS = 32768ull * 1024;  // 128 MB

  char* p = (char*)d_ws;
  auto take = [&](size_t bytes) {
    char* r = p; p += (bytes + 255) & ~(size_t)255; return r;
  };
  unsigned* Apk  = (unsigned*)take(APK_DW * 4);
  unsigned* Wpk  = (unsigned*)take(6 * WPK_DW * 4);
  unsigned* Upk  = (unsigned*)take(6 * UPK_DW * 4);
  float*    zf   = (float*)   take(Z_FLOATS * 4);
  float*    zb   = (float*)   take(Z_FLOATS * 4);
  float*    seqA = (float*)   take(SEQ_FLOATS * 4);
  float*    Pw   = (float*)   take(16384 * 4);
  float*    pbias= (float*)   take(64 * 4);
  unsigned* Ppk  = (unsigned*)take(32ull * 256 * 4);
  if ((size_t)(p - (char*)d_ws) > ws_size) return;   // workspace too small

  // ---- pack all weights into WMMA B-fragment layout (one pass) ----------
  for (int l = 0; l < 3; ++l) {
    pack_b_kernel<<<4096, 256, 0, stream>>>(Wf + (size_t)l * 1024 * 2048,
        Wpk + (size_t)(2 * l + 0) * WPK_DW, 32, 2048, (int)WPK_DW);
    pack_b_kernel<<<4096, 256, 0, stream>>>(Wb + (size_t)l * 1024 * 2048,
        Wpk + (size_t)(2 * l + 1) * WPK_DW, 32, 2048, (int)WPK_DW);
    pack_b_kernel<<<2048, 256, 0, stream>>>(Uf + (size_t)l * 512 * 2048,
        Upk + (size_t)(2 * l + 0) * UPK_DW, 16, 2048, (int)UPK_DW);
    pack_b_kernel<<<2048, 256, 0, stream>>>(Ub + (size_t)l * 512 * 2048,
        Upk + (size_t)(2 * l + 1) * UPK_DW, 16, 2048, (int)UPK_DW);
  }
  build_pw_kernel<<<64, 256, 0, stream>>>(Wp, Wbu, Wss3, Wss8,
                                          bp, bbu, bss3, bss8, Pw, pbias);
  pack_b_kernel<<<32, 256, 0, stream>>>(Pw, Ppk, 32, 16, 8192);

  // ---- 3 BiLSTM layers ---------------------------------------------------
  for (int l = 0; l < 3; ++l) {
    const float* src = (l == 0) ? x : seqA;
    pack_a_kernel<<<65536, 256, 0, stream>>>(src, Apk, 32, 1024, 0, (int)APK_DW);

    dim3 g(2048 / 8, 128 / 8);   // 256 x 16 blocks
    gemm_kernel<<<g, 256, 0, stream>>>((const u32x8*)Apk,
        (const u32x8*)(Wpk + (size_t)(2 * l + 0) * WPK_DW),
        bfv + (size_t)l * 2048, zf, 32, 128);
    gemm_kernel<<<g, 256, 0, stream>>>((const u32x8*)Apk,
        (const u32x8*)(Wpk + (size_t)(2 * l + 1) * WPK_DW),
        bbv + (size_t)l * 2048, zb, 32, 128);

    lstm_recur_kernel<<<2, 512, 0, stream>>>(zf, zb,
        (const u32x8*)(Upk + (size_t)(2 * l + 0) * UPK_DW),
        (const u32x8*)(Upk + (size_t)(2 * l + 1) * UPK_DW),
        xmask, seqA, T);
  }

  // ---- relu + fused 4-head projection ------------------------------------
  pack_a_kernel<<<65536, 256, 0, stream>>>(seqA, Apk, 32, 1024, 1, (int)APK_DW);
  proj_kernel<<<128, 256, 0, stream>>>((const u32x8*)Apk, (const u32x8*)Ppk,
                                       pbias, (float*)d_out);
}
